// DIFFBlock_90031104459369
// MI455X (gfx1250) — matrix operand order repaired and checked by reading, not compiled, not run on hardware
//
#include <hip/hip_runtime.h>
#include <hip/hip_bf16.h>
#include <math.h>

// Problem constants (from reference)
#define BD    4
#define SEQ   1024
#define DIM   768
#define NH    12
#define HDIM  64
#define MLPD  3072
#define ROWS  (BD * SEQ)   // 4096 token rows
#define LAMBDA_INIT 0.1f

typedef __attribute__((ext_vector_type(16))) __bf16 v16bf;
typedef __attribute__((ext_vector_type(8)))  __bf16 v8bf;
typedef __attribute__((ext_vector_type(8)))  float  v8f;
typedef __attribute__((ext_vector_type(4)))  int    v4i;

union V16 { v16bf v; v8bf h[2]; };

static __device__ __forceinline__ v8f wmma_bf16(v16bf a, v16bf b, v8f c) {
  // D(f32 16x16) = A(bf16 16x32) * B(bf16 32x16) + C
  return __builtin_amdgcn_wmma_f32_16x16x32_bf16(false, a, false, b, (short)0, c,
                                                 false, false);
}

static __device__ __forceinline__ v8f zero8() {
  v8f z = {0.f, 0.f, 0.f, 0.f, 0.f, 0.f, 0.f, 0.f};
  return z;
}

// ---- CDNA5 async global->LDS staging (direct-to-LDS DMA, ASYNCcnt-tracked) --
static __device__ __forceinline__ void async_copy16(const __bf16* g, __bf16* l) {
#if __has_builtin(__builtin_amdgcn_global_load_async_to_lds_b128)
  __builtin_amdgcn_global_load_async_to_lds_b128(
      (__attribute__((address_space(1))) v4i*)g,
      (__attribute__((address_space(3))) v4i*)l, 0, 0);
#else
  *(v8bf*)l = *(const v8bf*)g;  // sync fallback through VGPRs
#endif
}

static __device__ __forceinline__ void wait_async() {
#if __has_builtin(__builtin_amdgcn_s_wait_asynccnt)
  __builtin_amdgcn_s_wait_asynccnt(0);
#else
  asm volatile("s_wait_asynccnt 0" ::: "memory");
#endif
}

// reduce across the 16 lanes of each half-wave (rows of a WMMA C tile)
static __device__ __forceinline__ float halfmax(float v) {
#pragma unroll
  for (int m = 1; m < 16; m <<= 1) v = fmaxf(v, __shfl_xor(v, m, 32));
  return v;
}
static __device__ __forceinline__ float halfsum(float v) {
#pragma unroll
  for (int m = 1; m < 16; m <<= 1) v += __shfl_xor(v, m, 32);
  return v;
}

// ---------------------------------------------------------------- cvt f32->bf16
__global__ __launch_bounds__(256) void cvt_f32_bf16(const float* __restrict__ s,
                                                    __bf16* __restrict__ d, int n) {
  int i = blockIdx.x * 256 + threadIdx.x;
  if (i < n) d[i] = (__bf16)s[i];
}

// ------------------------------------------------- LayerNorm / RMSNorm -> bf16
__global__ __launch_bounds__(256) void norm_kernel(const float* __restrict__ X,
                                                   const float* __restrict__ w,
                                                   const float* __restrict__ bias,
                                                   __bf16* __restrict__ Y,
                                                   float eps, int is_rms) {
  const int row = blockIdx.x, tid = threadIdx.x;
  const float* xr = X + (size_t)row * DIM;
  float v[3], s = 0.f, ss = 0.f;
#pragma unroll
  for (int i = 0; i < 3; ++i) {
    v[i] = xr[tid + i * 256];
    s += v[i];
    ss += v[i] * v[i];
  }
#pragma unroll
  for (int m = 1; m < 32; m <<= 1) {
    s += __shfl_xor(s, m, 32);
    ss += __shfl_xor(ss, m, 32);
  }
  __shared__ float r0[8], r1[8];
  if ((tid & 31) == 0) { r0[tid >> 5] = s; r1[tid >> 5] = ss; }
  __syncthreads();
  s = 0.f; ss = 0.f;
#pragma unroll
  for (int i = 0; i < 8; ++i) { s += r0[i]; ss += r1[i]; }
  const float mean = is_rms ? 0.f : s * (1.f / DIM);
  const float var  = ss * (1.f / DIM) - mean * mean;
  const float inv  = rsqrtf(var + eps);
#pragma unroll
  for (int i = 0; i < 3; ++i) {
    const int c = tid + i * 256;
    float y = (v[i] - mean) * inv * w[c] + (bias ? bias[c] : 0.f);
    Y[(size_t)row * DIM + c] = (__bf16)y;
  }
}

// ------------------------------------------------------------- bf16 WMMA GEMM
// C[M,Nc] = A[M,K] @ Bw[Nc,K]^T   (both bf16, row-major, K contiguous)
// Double-buffered LDS, async direct-to-LDS staging.
// epi: 0 = scatter q/k/v bf16 (v transposed [B,H,HD,N])
//      1 = scatter q/k only
//      2 = outf = acc + bias[n] + res[m*Nc+n]        (fp32)
//      3 = outb = gelu(acc + bias[n])                (bf16)
__global__ __launch_bounds__(256) void gemm_bf16(
    const __bf16* __restrict__ A, const __bf16* __restrict__ Bw,
    int M, int Nc, int K, int epi,
    const float* __restrict__ bias, const float* __restrict__ res,
    float* __restrict__ outf, __bf16* __restrict__ outb,
    __bf16* __restrict__ oq, __bf16* __restrict__ ok, __bf16* __restrict__ ov) {
  __shared__ __align__(16) __bf16 sA[2][128 * 40];  // 32 data + 8 pad halfs/row
  __shared__ __align__(16) __bf16 sB[2][128 * 40];

  const int tid = threadIdx.x;
  const int lane = tid & 31, wid = tid >> 5;
  const int blockM = blockIdx.y * 128, blockN = blockIdx.x * 128;
  const int mw = (wid & 1) * 64, nw = (wid >> 1) * 32;

  v8f acc[4][2];
#pragma unroll
  for (int mi = 0; mi < 4; ++mi)
#pragma unroll
    for (int ni = 0; ni < 2; ++ni) acc[mi][ni] = zero8();

  const int lrow = tid >> 1, lhalf = tid & 1;  // 128 rows x 2 16-elem chunks
  const __bf16* gA = A + (size_t)(blockM + lrow) * K + lhalf * 16;
  const __bf16* gB = Bw + (size_t)(blockN + lrow) * K + lhalf * 16;
  const int lds_off = lrow * 40 + lhalf * 16;

  const int am = lane & 15, ak = (lane >> 4) * 8;   // A-frag lane mapping
  const int bn = lane & 15, bk = (lane >> 4) * 8;   // B-frag lane mapping

  auto stage = [&](int buf, int k0) {
    async_copy16(gA + k0,     sA[buf] + lds_off);
    async_copy16(gA + k0 + 8, sA[buf] + lds_off + 8);
    async_copy16(gB + k0,     sB[buf] + lds_off);
    async_copy16(gB + k0 + 8, sB[buf] + lds_off + 8);
    __builtin_prefetch(gA + k0 + 64, 0, 3);
    __builtin_prefetch(gB + k0 + 64, 0, 3);
  };

  const int nk = K / 32;
  stage(0, 0);
  for (int t = 0; t < nk; ++t) {
    wait_async();      // tile t resident in LDS (this wave's DMAs done)
    __syncthreads();   // all waves' tile-t data published; buf (t+1)&1 free
    if (t + 1 < nk) stage((t + 1) & 1, (t + 1) * 32);
    const __bf16* cA = sA[t & 1];
    const __bf16* cB = sB[t & 1];

    V16 af[4], bfr[2];
#pragma unroll
    for (int mi = 0; mi < 4; ++mi) {
      const __bf16* p = cA + (mw + mi * 16 + am) * 40 + ak;
      af[mi].h[0] = *(const v8bf*)p;
      af[mi].h[1] = *(const v8bf*)(p + 16);
    }
#pragma unroll
    for (int ni = 0; ni < 2; ++ni) {
      const __bf16* p = cB + (nw + ni * 16 + bn) * 40 + bk;
      bfr[ni].h[0] = *(const v8bf*)p;
      bfr[ni].h[1] = *(const v8bf*)(p + 16);
    }
#pragma unroll
    for (int mi = 0; mi < 4; ++mi)
#pragma unroll
      for (int ni = 0; ni < 2; ++ni)
        acc[mi][ni] = wmma_bf16(af[mi].v, bfr[ni].v, acc[mi][ni]);
    __syncthreads();   // done reading buf t&1 before it is restaged
  }

  // epilogue (C layout: VGPR r -> row r + 8*(lane>=16), col = lane&15)
  const int cn = lane & 15, ch = lane >> 4;
#pragma unroll
  for (int mi = 0; mi < 4; ++mi) {
#pragma unroll
    for (int ni = 0; ni < 2; ++ni) {
#pragma unroll
      for (int r = 0; r < 8; ++r) {
        const int m = blockM + mw + mi * 16 + r + 8 * ch;
        const int n = blockN + nw + ni * 16 + cn;
        const float val = acc[mi][ni][r];
        if (epi <= 1) {
          const int part = n / DIM;        // 0=q 1=k 2=v
          const int within = n - part * DIM;
          const int head = within >> 6;
          const int d = within & 63;
          const int b = m >> 10;
          const int sq = m & 1023;
          const size_t qk = (((size_t)b * NH + head) * SEQ + sq) * HDIM + d;
          if (part == 0)       oq[qk] = (__bf16)val;
          else if (part == 1)  ok[qk] = (__bf16)val;
          else if (epi == 0)
            ov[(((size_t)b * NH + head) * HDIM + d) * SEQ + sq] = (__bf16)val;
        } else if (epi == 2) {
          const size_t idx = (size_t)m * Nc + n;
          outf[idx] = val + bias[n] + res[idx];
        } else {
          const float xg = val + bias[n];
          const float g = 0.5f * xg * (1.0f + erff(xg * 0.70710678118f));
          outb[(size_t)m * Nc + n] = (__bf16)g;
        }
      }
    }
  }
}

// ---------------------------------------------- fused differential attention
// One flash-attention step over a 32-key block for one branch.
static __device__ __forceinline__ void flash_step(
    int j, int lane, const V16 qf[2], const __bf16* __restrict__ Kp,
    const __bf16* __restrict__ Vtp, __bf16* sPw, float* mrun, float* lrun,
    v8f* Oa) {
  const int kn = lane & 15, kb = (lane >> 4) * 8;
  v8f s[2];
#pragma unroll
  for (int kt = 0; kt < 2; ++kt) {
    v8f a = zero8();
    const __bf16* krow = Kp + (size_t)(j * 32 + kt * 16 + kn) * HDIM;
#pragma unroll
    for (int kc = 0; kc < 2; ++kc) {
      V16 kf;
      kf.h[0] = *(const v8bf*)(krow + kc * 32 + kb);
      kf.h[1] = *(const v8bf*)(krow + kc * 32 + kb + 16);
      a = wmma_bf16(qf[kc].v, kf.v, a);
    }
    s[kt] = a;
  }
  const float sc = 0.125f;  // HD^-0.5
  const int ch = lane >> 4, cn = lane & 15;
#pragma unroll
  for (int r = 0; r < 8; ++r) {
    const float rm = halfmax(fmaxf(s[0][r], s[1][r]) * sc);
    const float mnew = fmaxf(mrun[r], rm);
    const float corr = __expf(mrun[r] - mnew);
    mrun[r] = mnew;
    const float p0 = __expf(s[0][r] * sc - mnew);
    const float p1 = __expf(s[1][r] * sc - mnew);
    lrun[r] = lrun[r] * corr + halfsum(p0 + p1);
#pragma unroll
    for (int t = 0; t < 4; ++t) Oa[t][r] *= corr;
    // C layout -> A layout staging in LDS (per-wave region; in-order DS ops)
    const int mrow = r + 8 * ch;
    sPw[mrow * 40 + cn]      = (__bf16)p0;
    sPw[mrow * 40 + 16 + cn] = (__bf16)p1;
  }
  // reload P as 16x32 bf16 A fragment
  V16 pf;
  const int pm = lane & 15, pk = (lane >> 4) * 8;
  pf.h[0] = *(const v8bf*)(sPw + pm * 40 + pk);
  pf.h[1] = *(const v8bf*)(sPw + pm * 40 + pk + 16);
  // O += P @ V  (V transposed: keys contiguous)
#pragma unroll
  for (int t = 0; t < 4; ++t) {
    V16 vf;
    const __bf16* vp = Vtp + (size_t)(t * 16 + kn) * SEQ + j * 32 + kb;
    vf.h[0] = *(const v8bf*)vp;
    vf.h[1] = *(const v8bf*)(vp + 16);
    Oa[t] = wmma_bf16(pf.v, vf.v, Oa[t]);
  }
}

__global__ __launch_bounds__(256) void attn_diff(
    const __bf16* __restrict__ Q1, const __bf16* __restrict__ K1,
    const __bf16* __restrict__ Q2, const __bf16* __restrict__ K2,
    const __bf16* __restrict__ Vt, const float* __restrict__ lam1,
    const float* __restrict__ lam2, float* __restrict__ Out) {
  __shared__ __align__(16) __bf16 sP[8 * 16 * 40];
  const int tid = threadIdx.x, lane = tid & 31, wid = tid >> 5;
  const int bh = blockIdx.x, b = bh / NH, h = bh % NH;
  const int q0 = (blockIdx.y * 8 + wid) * 16;
  __bf16* sPw = sP + wid * 16 * 40;

  const size_t baseQK = (size_t)bh * SEQ * HDIM;
  const size_t baseV = (size_t)bh * HDIM * SEQ;

  const int am = lane & 15, ak = (lane >> 4) * 8;
  V16 q1f[2], q2f[2];
#pragma unroll
  for (int kc = 0; kc < 2; ++kc) {
    const __bf16* p1 = Q1 + baseQK + (size_t)(q0 + am) * HDIM + kc * 32 + ak;
    q1f[kc].h[0] = *(const v8bf*)p1;
    q1f[kc].h[1] = *(const v8bf*)(p1 + 16);
    const __bf16* p2 = Q2 + baseQK + (size_t)(q0 + am) * HDIM + kc * 32 + ak;
    q2f[kc].h[0] = *(const v8bf*)p2;
    q2f[kc].h[1] = *(const v8bf*)(p2 + 16);
  }

  float m1[8], l1[8], m2[8], l2[8];
  v8f O1[4], O2[4];
#pragma unroll
  for (int r = 0; r < 8; ++r) { m1[r] = -3.0e38f; m2[r] = -3.0e38f; l1[r] = 0.f; l2[r] = 0.f; }
#pragma unroll
  for (int t = 0; t < 4; ++t) { O1[t] = zero8(); O2[t] = zero8(); }

  for (int j = 0; j < SEQ / 32; ++j) {
    flash_step(j, lane, q1f, K1 + baseQK, Vt + baseV, sPw, m1, l1, O1);
    flash_step(j, lane, q2f, K2 + baseQK, Vt + baseV, sPw, m2, l2, O2);
  }

  const float lam = lam1[h] - lam2[h] + LAMBDA_INIT;
  const int ch = lane >> 4, cn = lane & 15;
#pragma unroll
  for (int r = 0; r < 8; ++r) {
    const float i1 = 1.f / l1[r], i2 = 1.f / l2[r];
    const int q = q0 + r + 8 * ch;
#pragma unroll
    for (int t = 0; t < 4; ++t) {
      const float val = O1[t][r] * i1 - lam * (O2[t][r] * i2);
      Out[((size_t)b * SEQ + q) * DIM + h * HDIM + t * 16 + cn] = val;
    }
  }
}

// -------------------------------------------------------------------- launch
extern "C" void kernel_launch(void* const* d_in, const int* in_sizes, int n_in,
                              void* d_out, int out_size, void* d_ws,
                              size_t ws_size, hipStream_t stream) {
  (void)in_sizes; (void)n_in; (void)out_size; (void)ws_size;
  const float* x      = (const float*)d_in[0];
  const float* ln1_w  = (const float*)d_in[2];
  const float* ln1_b  = (const float*)d_in[3];
  const float* qkv1_w = (const float*)d_in[4];
  const float* qkv2_w = (const float*)d_in[5];
  const float* proj_w = (const float*)d_in[6];
  const float* proj_b = (const float*)d_in[7];
  const float* rms_w  = (const float*)d_in[8];
  const float* lam1   = (const float*)d_in[9];
  const float* lam2   = (const float*)d_in[10];
  const float* ln2_w  = (const float*)d_in[11];
  const float* ln2_b  = (const float*)d_in[12];
  const float* fc1_w  = (const float*)d_in[13];
  const float* fc1_b  = (const float*)d_in[14];
  const float* fc2_w  = (const float*)d_in[15];
  const float* fc2_b  = (const float*)d_in[16];

  char* ws = (char*)d_ws;
  size_t off = 0;
  auto alloc = [&](size_t bytes) -> void* {
    off = (off + 255) & ~(size_t)255;
    void* p = (void*)(ws + off);
    off += bytes;
    return p;
  };

  const size_t nQKVW = (size_t)3 * DIM * DIM;
  const size_t nHDN  = (size_t)BD * NH * SEQ * HDIM;
  __bf16* wq1 = (__bf16*)alloc(nQKVW * 2);
  __bf16* wq2 = (__bf16*)alloc(nQKVW * 2);
  __bf16* wpj = (__bf16*)alloc((size_t)DIM * DIM * 2);
  __bf16* wf1 = (__bf16*)alloc((size_t)MLPD * DIM * 2);
  __bf16* wf2 = (__bf16*)alloc((size_t)DIM * MLPD * 2);
  __bf16* h1  = (__bf16*)alloc((size_t)ROWS * DIM * 2);
  __bf16* q1  = (__bf16*)alloc(nHDN * 2);
  __bf16* k1  = (__bf16*)alloc(nHDN * 2);
  __bf16* v1t = (__bf16*)alloc(nHDN * 2);
  __bf16* q2  = (__bf16*)alloc(nHDN * 2);
  __bf16* k2  = (__bf16*)alloc(nHDN * 2);
  float*  Of  = (float*)alloc((size_t)ROWS * DIM * 4);
  __bf16* onm = (__bf16*)alloc((size_t)ROWS * DIM * 2);
  float*  x1  = (float*)alloc((size_t)ROWS * DIM * 4);
  __bf16* h2  = (__bf16*)alloc((size_t)ROWS * DIM * 2);
  __bf16* mm1 = (__bf16*)alloc((size_t)ROWS * MLPD * 2);

  int n;
  n = (int)nQKVW;
  cvt_f32_bf16<<<(n + 255) / 256, 256, 0, stream>>>(qkv1_w, wq1, n);
  cvt_f32_bf16<<<(n + 255) / 256, 256, 0, stream>>>(qkv2_w, wq2, n);
  n = DIM * DIM;
  cvt_f32_bf16<<<(n + 255) / 256, 256, 0, stream>>>(proj_w, wpj, n);
  n = MLPD * DIM;
  cvt_f32_bf16<<<(n + 255) / 256, 256, 0, stream>>>(fc1_w, wf1, n);
  cvt_f32_bf16<<<(n + 255) / 256, 256, 0, stream>>>(fc2_w, wf2, n);

  // h = LN1(x) -> bf16
  norm_kernel<<<ROWS, 256, 0, stream>>>(x, ln1_w, ln1_b, h1, 1e-5f, 0);

  // qkv projections (scatter into per-head bf16 tensors, V transposed)
  gemm_bf16<<<dim3((3 * DIM) / 128, ROWS / 128), 256, 0, stream>>>(
      h1, wq1, ROWS, 3 * DIM, DIM, 0, nullptr, nullptr, nullptr, nullptr, q1, k1, v1t);
  gemm_bf16<<<dim3((3 * DIM) / 128, ROWS / 128), 256, 0, stream>>>(
      h1, wq2, ROWS, 3 * DIM, DIM, 1, nullptr, nullptr, nullptr, nullptr, q2, k2, nullptr);

  // differential flash attention
  attn_diff<<<dim3(BD * NH, SEQ / 128), 256, 0, stream>>>(q1, k1, q2, k2, v1t,
                                                          lam1, lam2, Of);

  // rmsnorm -> bf16
  norm_kernel<<<ROWS, 256, 0, stream>>>(Of, rms_w, nullptr, onm, 1e-6f, 1);

  // x1 = x + o @ proj_w^T + proj_b
  gemm_bf16<<<dim3(DIM / 128, ROWS / 128), 256, 0, stream>>>(
      onm, wpj, ROWS, DIM, DIM, 2, proj_b, x, x1, nullptr, nullptr, nullptr, nullptr);

  // h2 = LN2(x1) -> bf16
  norm_kernel<<<ROWS, 256, 0, stream>>>(x1, ln2_w, ln2_b, h2, 1e-5f, 0);

  // m1 = gelu(h2 @ fc1^T + b1) -> bf16
  gemm_bf16<<<dim3(MLPD / 128, ROWS / 128), 256, 0, stream>>>(
      h2, wf1, ROWS, MLPD, DIM, 3, fc1_b, nullptr, nullptr, mm1, nullptr, nullptr, nullptr);

  // out = x1 + m1 @ fc2^T + b2
  gemm_bf16<<<dim3(DIM / 128, ROWS / 128), 256, 0, stream>>>(
      mm1, wf2, ROWS, DIM, MLPD, 2, fc2_b, x1, (float*)d_out, nullptr, nullptr, nullptr, nullptr);
}
